// CrossAttention_37546604101755
// MI455X (gfx1250) — matrix operand order repaired and checked by reading, not compiled, not run on hardware
//
#include <hip/hip_runtime.h>
#include <hip/hip_bf16.h>

typedef __attribute__((ext_vector_type(16))) _Float16 v16h;
typedef __attribute__((ext_vector_type(4)))  _Float16 v4h;
typedef __attribute__((ext_vector_type(8)))  float    v8f;

#define CS 1024
#define NH 16
#define HD 64
#define CZ 128

// ---------------------------------------------------------------- f32 -> f16 (x4 per thread)
__global__ void cvt_f32_f16(const float* __restrict__ src, _Float16* __restrict__ dst, int n) {
    int i = (blockIdx.x * blockDim.x + threadIdx.x) * 4;
    if (i < n) {
        float4 v = *(const float4*)(src + i);
        v4h o = { (_Float16)v.x, (_Float16)v.y, (_Float16)v.z, (_Float16)v.w };
        *(v4h*)(dst + i) = o;
    }
}

// ------------------------------------------------- C = A(1024x1024) @ W^T, WMMA
// one wave per 16x16 output tile; A,W f16 (L2-resident); optional bias-vector,
// sigmoid epilogue, f16 and/or f32 output.
__global__ void gemm_awT(const _Float16* __restrict__ A, const _Float16* __restrict__ W,
                         const float* __restrict__ bvec, _Float16* __restrict__ outH,
                         float* __restrict__ outF, int sigm) {
    const int lane = threadIdx.x & 31;
    const int g    = lane >> 4;
    const int nn   = lane & 15;
    const int wave = blockIdx.x * (blockDim.x >> 5) + (threadIdx.x >> 5);
    const int tm   = wave >> 6;   // 0..63
    const int tn   = wave & 63;   // 0..63

    const _Float16* arow = A + (size_t)(tm * 16 + nn) * CS;  // A-frag row m = lane&15
    const _Float16* brow = W + (size_t)(tn * 16 + nn) * CS;  // B-frag col n = lane&15

    v8f acc = {};
    for (int kb = 0; kb < CS; kb += 32) {
        v16h a, b;
        ((float4*)&a)[0] = *(const float4*)(arow + kb + g * 8);        // K = g*8 .. +7
        ((float4*)&a)[1] = *(const float4*)(arow + kb + 16 + g * 8);   // K = 16+g*8 ..
        ((float4*)&b)[0] = *(const float4*)(brow + kb + g * 8);        // K = g*8 .. +7
        ((float4*)&b)[1] = *(const float4*)(brow + kb + 16 + g * 8);
        acc = __builtin_amdgcn_wmma_f32_16x16x32_f16(false, a, false, b,
                                                     (short)0, acc, false, false);
    }

    const int col  = tn * 16 + nn;
    const float bb = bvec ? bvec[col] : 0.0f;
#pragma unroll
    for (int r = 0; r < 8; ++r) {
        int   row = tm * 16 + 8 * g + r;
        float v   = acc[r] + bb;
        if (sigm) v = 1.0f / (1.0f + __expf(-v));
        if (outH) outH[(size_t)row * CS + col] = (_Float16)v;
        if (outF) outF[(size_t)row * CS + col] = v;
    }
}

// ------------------------------------------------- v[j][c] -> vt[c][j]
__global__ void transpose_cs(const _Float16* __restrict__ src, _Float16* __restrict__ dst) {
    int idx = blockIdx.x * blockDim.x + threadIdx.x;  // c*1024 + j
    int c = idx >> 10, j = idx & 1023;
    dst[idx] = src[j * CS + c];
}

// ------------------------------------------------- z[p][h] = sum_c bias[p][c]*Wz[c][h]
// Streams the 512MB bias tensor exactly once via CDNA5 async global->LDS copies:
// each wave cooperatively stages 32 pairs x 128B per c-chunk (512B per async
// instruction), then computes the 16-head projection out of padded LDS.
#define ZP_PAD 36  // floats per pair in stage (32 data + 4 pad -> bank spread)

__global__ void zproj(const float* __restrict__ bias, const float* __restrict__ Wz,
                      _Float16* __restrict__ z16) {
    __shared__ float sW[CZ * NH];                             // 8 KB, Wz cache
    __shared__ __align__(16) float stage[8][32 * ZP_PAD];     // 8 waves x 4.5 KB

    for (int t = threadIdx.x; t < CZ * NH; t += blockDim.x) sW[t] = Wz[t];
    __syncthreads();

    const int    lane = threadIdx.x & 31;
    const int    w    = threadIdx.x >> 5;
    const size_t p0   = ((size_t)blockIdx.x * 8 + w) * 32;    // first (i*1024+j) pair

    float acc[NH];
#pragma unroll
    for (int h = 0; h < NH; ++h) acc[h] = 0.0f;

    for (int cc = 0; cc < 4; ++cc) {
        // DS reads of previous chunk must land before async engine overwrites stage
        asm volatile("s_wait_dscnt 0x0" ::: "memory");
        // async-stage: 32 pairs x 32 floats; flat float4 index e = pair*8 + o4
#pragma unroll
        for (int q = 0; q < 8; ++q) {
            int pl = (q * 32 + lane) >> 3;
            int o4 = (q * 32 + lane) & 7;
            const float*       gp = bias + (p0 + pl) * CZ + cc * 32 + o4 * 4;
            unsigned           lds = (unsigned)(size_t)(&stage[w][pl * ZP_PAD]) +
                                     (unsigned)(o4 * 16);
            unsigned long long ga  = (unsigned long long)(size_t)gp;
            asm volatile("global_load_async_to_lds_b128 %0, %1, off"
                         :: "v"(lds), "v"(ga) : "memory");
        }
        asm volatile("s_wait_asynccnt 0x0" ::: "memory");

        const float* sp = &stage[w][lane * ZP_PAD];  // this lane's pair, 128B
#pragma unroll
        for (int t = 0; t < 8; ++t) {
            float4 bv = ((const float4*)sp)[t];
            int    c  = cc * 32 + t * 4;
#pragma unroll
            for (int h = 0; h < NH; ++h) {
                acc[h] += bv.x * sW[(c + 0) * NH + h] + bv.y * sW[(c + 1) * NH + h] +
                          bv.z * sW[(c + 2) * NH + h] + bv.w * sW[(c + 3) * NH + h];
            }
        }
    }

    _Float16* zp = z16 + (p0 + lane) * NH;
#pragma unroll
    for (int h = 0; h < NH; ++h) zp[h] = (_Float16)acc[h];
}

// ------------------------------------------------- flash attention + gating
// one wave per (head, 16-row i-tile); j-loop step 32; online softmax.
__global__ void attn_kernel(const _Float16* __restrict__ q16, const _Float16* __restrict__ k16,
                            const _Float16* __restrict__ vt16, const _Float16* __restrict__ z16,
                            const float* __restrict__ mask, const float* __restrict__ g32,
                            _Float16* __restrict__ go16) {
    __shared__ __align__(16) _Float16 Pl[8][16 * 32];

    const int lane = threadIdx.x & 31;
    const int g    = lane >> 4;
    const int nn   = lane & 15;
    const int w    = threadIdx.x >> 5;
    const int wave = blockIdx.x * 8 + w;
    const int h    = wave & 15;
    const int it   = wave >> 4;  // 0..63

    // Q fragments: 16x64 split into two 16x32 A-fragments
    v16h qa0, qa1;
    {
        const _Float16* qr = q16 + (size_t)(it * 16 + nn) * CS + h * HD;
        ((float4*)&qa0)[0] = *(const float4*)(qr + 0  + g * 16);
        ((float4*)&qa0)[1] = *(const float4*)(qr + 8  + g * 16);
        ((float4*)&qa1)[0] = *(const float4*)(qr + 32 + g * 16);
        ((float4*)&qa1)[1] = *(const float4*)(qr + 40 + g * 16);
    }

    v8f o0 = {}, o1 = {}, o2 = {}, o3 = {};
    float mx[8], lsum[8];
#pragma unroll
    for (int r = 0; r < 8; ++r) { mx[r] = -3.0e38f; lsum[r] = 0.0f; }

    for (int jb = 0; jb < 1024; jb += 32) {
        // ---- scores: S[i][j] = q.k  (two 16-col j tiles, K=64 => 2 WMMAs each)
        v8f s0 = {}, s1 = {};
        {
            const _Float16* kr0 = k16 + (size_t)(jb + nn) * CS + h * HD;
            const _Float16* kr1 = k16 + (size_t)(jb + 16 + nn) * CS + h * HD;
            v16h b;
            ((float4*)&b)[0] = *(const float4*)(kr0 + g * 16);
            ((float4*)&b)[1] = *(const float4*)(kr0 + g * 16 + 8);
            s0 = __builtin_amdgcn_wmma_f32_16x16x32_f16(false, qa0, false, b, (short)0, s0, false, false);
            ((float4*)&b)[0] = *(const float4*)(kr0 + 32 + g * 16);
            ((float4*)&b)[1] = *(const float4*)(kr0 + 32 + g * 16 + 8);
            s0 = __builtin_amdgcn_wmma_f32_16x16x32_f16(false, qa1, false, b, (short)0, s0, false, false);
            ((float4*)&b)[0] = *(const float4*)(kr1 + g * 16);
            ((float4*)&b)[1] = *(const float4*)(kr1 + g * 16 + 8);
            s1 = __builtin_amdgcn_wmma_f32_16x16x32_f16(false, qa0, false, b, (short)0, s1, false, false);
            ((float4*)&b)[0] = *(const float4*)(kr1 + 32 + g * 16);
            ((float4*)&b)[1] = *(const float4*)(kr1 + 32 + g * 16 + 8);
            s1 = __builtin_amdgcn_wmma_f32_16x16x32_f16(false, qa1, false, b, (short)0, s1, false, false);
        }

        // ---- scale + pair-bias z + mask penalty
        const float pen0 = (1.0f - mask[jb + nn])      * -1.0e6f;
        const float pen1 = (1.0f - mask[jb + 16 + nn]) * -1.0e6f;
        float sc0[8], sc1[8];
#pragma unroll
        for (int r = 0; r < 8; ++r) {
            int   i  = it * 16 + 8 * g + r;
            float z0 = (float)z16[((size_t)i * 1024 + jb + nn) * NH + h];
            float z1 = (float)z16[((size_t)i * 1024 + jb + 16 + nn) * NH + h];
            sc0[r] = s0[r] * 0.125f + z0 + pen0;
            sc1[r] = s1[r] * 0.125f + z1 + pen1;
        }

        // ---- online softmax (row reductions across 16-lane groups)
#pragma unroll
        for (int r = 0; r < 8; ++r) {
            float rmax = fmaxf(sc0[r], sc1[r]);
#pragma unroll
            for (int off = 1; off < 16; off <<= 1)
                rmax = fmaxf(rmax, __shfl_xor(rmax, off, 32));
            float mnew  = fmaxf(mx[r], rmax);
            float alpha = __expf(mx[r] - mnew);
            float p0    = __expf(sc0[r] - mnew);
            float p1    = __expf(sc1[r] - mnew);
            float rs    = p0 + p1;
#pragma unroll
            for (int off = 1; off < 16; off <<= 1)
                rs += __shfl_xor(rs, off, 32);
            lsum[r] = lsum[r] * alpha + rs;
            mx[r]   = mnew;
            o0[r] *= alpha; o1[r] *= alpha; o2[r] *= alpha; o3[r] *= alpha;
            int m = 8 * g + r;  // C-layout -> LDS row-major P tile
            Pl[w][m * 32 + nn]      = (_Float16)p0;
            Pl[w][m * 32 + 16 + nn] = (_Float16)p1;
        }
        __syncthreads();

        // ---- P as A-fragment (16x32) from LDS
        v16h pa;
        ((float4*)&pa)[0] = *(const float4*)(&Pl[w][nn * 32 + g * 16]);
        ((float4*)&pa)[1] = *(const float4*)(&Pl[w][nn * 32 + g * 16 + 8]);

        // ---- o += P @ V : 4 d-chunks of 16 columns (V pre-transposed: vt[c][j])
        const _Float16* vb = vt16 + (size_t)(h * HD) * 1024 + jb + g * 16;
        v16h b;
        ((float4*)&b)[0] = *(const float4*)(vb + (size_t)(0 * 16 + nn) * 1024);
        ((float4*)&b)[1] = *(const float4*)(vb + (size_t)(0 * 16 + nn) * 1024 + 8);
        o0 = __builtin_amdgcn_wmma_f32_16x16x32_f16(false, pa, false, b, (short)0, o0, false, false);
        ((float4*)&b)[0] = *(const float4*)(vb + (size_t)(1 * 16 + nn) * 1024);
        ((float4*)&b)[1] = *(const float4*)(vb + (size_t)(1 * 16 + nn) * 1024 + 8);
        o1 = __builtin_amdgcn_wmma_f32_16x16x32_f16(false, pa, false, b, (short)0, o1, false, false);
        ((float4*)&b)[0] = *(const float4*)(vb + (size_t)(2 * 16 + nn) * 1024);
        ((float4*)&b)[1] = *(const float4*)(vb + (size_t)(2 * 16 + nn) * 1024 + 8);
        o2 = __builtin_amdgcn_wmma_f32_16x16x32_f16(false, pa, false, b, (short)0, o2, false, false);
        ((float4*)&b)[0] = *(const float4*)(vb + (size_t)(3 * 16 + nn) * 1024);
        ((float4*)&b)[1] = *(const float4*)(vb + (size_t)(3 * 16 + nn) * 1024 + 8);
        o3 = __builtin_amdgcn_wmma_f32_16x16x32_f16(false, pa, false, b, (short)0, o3, false, false);
        __syncthreads();
    }

    // ---- normalize, gate with g = sigmoid(s Wg^T), store f16
#pragma unroll
    for (int r = 0; r < 8; ++r) {
        int    i    = it * 16 + 8 * g + r;
        float  inv  = 1.0f / lsum[r];
        size_t base = (size_t)i * CS + h * HD + nn;
        float v;
        v = o0[r] * inv; go16[base + 0 ] = (_Float16)(v * g32[base + 0 ]);
        v = o1[r] * inv; go16[base + 16] = (_Float16)(v * g32[base + 16]);
        v = o2[r] * inv; go16[base + 32] = (_Float16)(v * g32[base + 32]);
        v = o3[r] * inv; go16[base + 48] = (_Float16)(v * g32[base + 48]);
    }
}

// ------------------------------------------------------------------ launcher
extern "C" void kernel_launch(void* const* d_in, const int* in_sizes, int n_in,
                              void* d_out, int out_size, void* d_ws, size_t ws_size,
                              hipStream_t stream) {
    (void)in_sizes; (void)n_in; (void)out_size; (void)ws_size;

    const float* s    = (const float*)d_in[0];
    const float* k_in = (const float*)d_in[1];
    const float* mask = (const float*)d_in[2];
    const float* bias = (const float*)d_in[3];
    const float* Wq   = (const float*)d_in[4];
    const float* bq   = (const float*)d_in[5];
    const float* Wk   = (const float*)d_in[6];
    const float* Wv   = (const float*)d_in[7];
    const float* Wg   = (const float*)d_in[8];
    const float* Wo   = (const float*)d_in[9];
    const float* Wz   = (const float*)d_in[10];
    // d_in[11] = multiplicity (== 1, no-op for B=1)
    float* out = (float*)d_out;

    char* ws = (char*)d_ws;
    const size_t MB = (size_t)1 << 20;
    _Float16* s16   = (_Float16*)(ws + 0 * MB);
    _Float16* kin16 = (_Float16*)(ws + 2 * MB);
    _Float16* Wq16  = (_Float16*)(ws + 4 * MB);
    _Float16* Wk16  = (_Float16*)(ws + 6 * MB);
    _Float16* Wv16  = (_Float16*)(ws + 8 * MB);
    _Float16* Wg16  = (_Float16*)(ws + 10 * MB);
    _Float16* Wo16  = (_Float16*)(ws + 12 * MB);
    _Float16* q16   = (_Float16*)(ws + 14 * MB);
    _Float16* k16   = (_Float16*)(ws + 16 * MB);
    _Float16* v16   = (_Float16*)(ws + 18 * MB);
    _Float16* vt16  = (_Float16*)(ws + 20 * MB);
    float*    g32   = (float*)   (ws + 22 * MB);   // 4 MB
    _Float16* go16  = (_Float16*)(ws + 26 * MB);   // 2 MB
    _Float16* z16   = (_Float16*)(ws + 28 * MB);   // 32 MB -> total 60 MB

    const int N1 = 1024 * 1024;
    const int CB = 256;

    // 1) f32 -> f16 (4 elements / thread)
    cvt_f32_f16<<<N1 / (CB * 4), CB, 0, stream>>>(s,    s16,   N1);
    cvt_f32_f16<<<N1 / (CB * 4), CB, 0, stream>>>(k_in, kin16, N1);
    cvt_f32_f16<<<N1 / (CB * 4), CB, 0, stream>>>(Wq,   Wq16,  N1);
    cvt_f32_f16<<<N1 / (CB * 4), CB, 0, stream>>>(Wk,   Wk16,  N1);
    cvt_f32_f16<<<N1 / (CB * 4), CB, 0, stream>>>(Wv,   Wv16,  N1);
    cvt_f32_f16<<<N1 / (CB * 4), CB, 0, stream>>>(Wg,   Wg16,  N1);
    cvt_f32_f16<<<N1 / (CB * 4), CB, 0, stream>>>(Wo,   Wo16,  N1);

    // 2) projections (WMMA): 4096 tiles / 8 waves per block = 512 blocks
    gemm_awT<<<512, 256, 0, stream>>>(s16,   Wq16, bq,      q16,     nullptr, 0);
    gemm_awT<<<512, 256, 0, stream>>>(kin16, Wk16, nullptr, k16,     nullptr, 0);
    gemm_awT<<<512, 256, 0, stream>>>(kin16, Wv16, nullptr, v16,     nullptr, 0);
    gemm_awT<<<512, 256, 0, stream>>>(s16,   Wg16, nullptr, nullptr, g32,     1);

    // 3) V transpose for PV fragments
    transpose_cs<<<N1 / CB, CB, 0, stream>>>(v16, vt16);

    // 4) pair-bias projection: 32768 waves, 32 pairs each -> 4096 blocks
    zproj<<<4096, 256, 0, stream>>>(bias, Wz, z16);

    // 5) attention + gating: 16 heads * 64 i-tiles = 1024 waves / 8 = 128 blocks
    attn_kernel<<<128, 256, 0, stream>>>(q16, k16, vt16, z16, mask, g32, go16);

    // 6) output projection -> f32
    gemm_awT<<<512, 256, 0, stream>>>(go16, Wo16, nullptr, nullptr, out, 0);
}
